// SoftNCutsLoss_v3_38611755991340
// MI455X (gfx1250) — compile-verified, matrix operand order checked by reading, not run on hardware
//
#include <hip/hip_runtime.h>

// SoftNCutsLoss for MI455X (gfx1250, wave32).
// Pass 1: 243-offset stencil (dist mask kills all but dx in {-1,0,1}),
//         branchless clamped loads, v_exp_f32 + FMA co-execution.
// Pass 2: 16x16x262144 f32 GEMM via V_WMMA_F32_16X16X4_F32, branchless frags.
// Pass 3: tiny reduction + final loss.

#define PDIM 64
#define P2   (PDIM * PDIM)
#define P3   (PDIM * PDIM * PDIM)   // 262144
#define NB   4
#define NK   4
#define RAD  4
#define EPSF 2.2204460492503131e-16f
#define WDX1 0.8290291181804004f    // exp(-3/16)

#define WAVES2 256
#define VPW    (P3 / WAVES2)        // 1024 voxels per wave in pass 2

typedef __attribute__((ext_vector_type(2)))  float    v2f;
typedef __attribute__((ext_vector_type(8)))  float    v8f;
typedef __attribute__((ext_vector_type(16))) _Float16 v16h;

__device__ __forceinline__ int clampi(int v, int lo, int hi)
{
    return v < lo ? lo : (v > hi ? hi : v);
}

// ---------------------------------------------------------------------------
// Pass 1: per-voxel stencil. One thread per voxel; loops b' (4) x dz,dy in
// [-4,4] x dx in {-1,0,1}. Out-of-range reads replicate jnp.pad(...,EPS):
// addresses are clamped (always-legal unconditional loads), values selected.
// ---------------------------------------------------------------------------
__global__ __launch_bounds__(256)
void sncut_stencil(const float* __restrict__ batch,
                   const float* __restrict__ preds,
                   float* __restrict__ num_s,
                   float* __restrict__ den_s)
{
    const int x = threadIdx.x & 63;
    const int y = (blockIdx.y << 2) + (threadIdx.x >> 6);
    const int z = blockIdx.x;
    const int v = z * P2 + y * PDIM + x;

    float num0 = 0.f, num1 = 0.f, num2 = 0.f, num3 = 0.f, den = 0.f;

    for (int b = 0; b < NB; ++b) {
        const float* __restrict__ bat = batch + b * P3;
        const float* __restrict__ prd = preds + (b * NK) * P3;
        const float Iv = bat[v];
        for (int dz = -RAD; dz <= RAD; ++dz) {
            const int zz  = z + dz;
            const int zzc = clampi(zz, 0, PDIM - 1);
            const bool zin = (unsigned)zz < (unsigned)PDIM;
            for (int dy = -RAD; dy <= RAD; ++dy) {
                const int yy  = y + dy;
                const int yyc = clampi(yy, 0, PDIM - 1);
                const bool zyin = zin && ((unsigned)yy < (unsigned)PDIM);
                const int rowc = zzc * P2 + yyc * PDIM;
#pragma unroll
                for (int dx = -1; dx <= 1; ++dx) {
                    const int xx  = x + dx;
                    const int xxc = clampi(xx, 0, PDIM - 1);
                    const bool inb = zyin && ((unsigned)xx < (unsigned)PDIM);
                    const int sidx = rowc + xxc;
                    // Unconditional loads from clamped (always valid) addr,
                    // then value-select EPS for padded positions.
                    float img = bat[sidx];
                    float p0  = prd[sidx];
                    float p1  = prd[sidx + P3];
                    float p2  = prd[sidx + 2 * P3];
                    float p3  = prd[sidx + 3 * P3];
                    img = inb ? img : EPSF;
                    p0  = inb ? p0  : EPSF;
                    p1  = inb ? p1  : EPSF;
                    p2  = inb ? p2  : EPSF;
                    p3  = inb ? p3  : EPSF;
                    const float d = img - Iv;
                    // w = exp(-d^2 / SIGMA_I^2) * spatial_weight(dx)
                    const float w = (dx == 0 ? 1.0f : WDX1) *
                                    __expf(d * d * (-0.01f));
                    den += w;
                    num0 = fmaf(w, p0, num0);
                    num1 = fmaf(w, p1, num1);
                    num2 = fmaf(w, p2, num2);
                    num3 = fmaf(w, p3, num3);
                }
            }
        }
    }

    num_s[v]          = num0;
    num_s[P3 + v]     = num1;
    num_s[2 * P3 + v] = num2;
    num_s[3 * P3 + v] = num3;
    den_s[v]          = den;
}

// ---------------------------------------------------------------------------
// Pass 2: D[m,n] = sum_v A[m,v] * B[v,n], m = b*4+k (16 rows of preds),
// n<4 -> num_s channel n, n==4 -> den_s, n>4 -> 0. 256 waves, each owns a
// contiguous 1024-voxel K-chunk and accumulates a 16x16 f32 tile with
// V_WMMA_F32_16X16X4_F32. Per-lane base pointers are hoisted; the n>4
// columns are zeroed with cndmask (no divergent branches in the loop).
// ---------------------------------------------------------------------------
__global__ __launch_bounds__(128)
void sncut_wmma(const float* __restrict__ preds,
                const float* __restrict__ num_s,
                const float* __restrict__ den_s,
                float* __restrict__ partials)
{
    const int wave = blockIdx.x * 4 + (threadIdx.x >> 5);
    const int lane = threadIdx.x & 31;
    const int hi   = lane >> 4;     // half-wave select
    const int mn   = lane & 15;     // M for A-frag, N for B-frag

    v8f c = {};

#if !defined(__HIP_DEVICE_COMPILE__)
    (void)preds; (void)num_s; (void)den_s; (void)wave; (void)hi; (void)mn;
#elif __has_builtin(__builtin_amdgcn_wmma_f32_16x16x4_f32)
    // A 16x4 f32 frag: lanes 0-15 hold K=0 (reg0) / K=1 (reg1),
    //                  lanes 16-31 hold K=2 / K=3.  B 4x16 is symmetric.
    // Both K-values per lane are adjacent in memory -> one b64 load each.
    const float* __restrict__ aptr = preds + mn * P3 + 2 * hi;
    const float* __restrict__ bptr =
        ((mn < NK) ? (num_s + mn * P3) : den_s) + 2 * hi;   // always valid
    const bool bon = (mn <= NK);
    const int vbase = wave * VPW;
#pragma unroll 4
    for (int t = 0; t < VPW; t += 4) {
        const int v0 = vbase + t;
        v2f a    = *(const v2f*)(aptr + v0);
        v2f braw = *(const v2f*)(bptr + v0);
        v2f bm;
        bm.x = bon ? braw.x : 0.f;
        bm.y = bon ? braw.y : 0.f;
        c = __builtin_amdgcn_wmma_f32_16x16x4_f32(
                /*neg_a=*/false, a, /*neg_b=*/false, bm,
                /*c_mod=*/(short)0, c, /*reuse_a=*/false, /*reuse_b=*/false);
    }
#else
    // Fallback: codegen-confirmed f16 WMMA (K=32 per instruction).
    const int vbase = wave * VPW;
    for (int t = 0; t < VPW; t += 32) {
        const int v0 = vbase + t;
        v16h a, bmat;
#pragma unroll
        for (int r = 0; r < 8; ++r) {
#pragma unroll
            for (int h = 0; h < 2; ++h) {
                const int K = 2 * (r & 3) + h + ((r & 4) ? 16 : 0) + (hi ? 8 : 0);
                const int vv = v0 + K;
                float bv = (mn < NK) ? num_s[mn * P3 + vv]
                                     : ((mn == NK) ? den_s[vv] : 0.f);
                a[2 * r + h]    = (_Float16)preds[mn * P3 + vv];
                bmat[2 * r + h] = (_Float16)bv;
            }
        }
        c = __builtin_amdgcn_wmma_f32_16x16x32_f16(
                false, a, false, bmat, (short)0, c, false, false);
    }
#endif

    // C/D layout: reg r, lanes 0-15 -> (M=r, N=lane); lanes 16-31 -> (M=r+8).
#pragma unroll
    for (int r = 0; r < 8; ++r)
        partials[wave * 256 + (r + 8 * hi) * 16 + mn] = c[r];
}

// ---------------------------------------------------------------------------
// Pass 3: reduce 256 partial tiles, then
//   out[b] = K - sum_k (D[b*4+k, k] + eps) / (D[b*4+k, 4] + eps)
// ---------------------------------------------------------------------------
__global__ __launch_bounds__(256)
void sncut_final(const float* __restrict__ partials, float* __restrict__ out)
{
    __shared__ float D[256];
    const int t = threadIdx.x;
    float s = 0.f;
    for (int w = 0; w < WAVES2; ++w)
        s += partials[w * 256 + t];
    D[t] = s;
    __syncthreads();
    if (t < NB) {
        float acc = 0.f;
        for (int k = 0; k < NK; ++k) {
            const int m = t * NK + k;
            acc += (D[m * 16 + k] + EPSF) / (D[m * 16 + NK] + EPSF);
        }
        out[t] = (float)NK - acc;
    }
}

// ---------------------------------------------------------------------------
extern "C" void kernel_launch(void* const* d_in, const int* in_sizes, int n_in,
                              void* d_out, int out_size, void* d_ws, size_t ws_size,
                              hipStream_t stream)
{
    (void)in_sizes; (void)n_in; (void)out_size; (void)ws_size;
    const float* batch = (const float*)d_in[0];   // [4,1,64,64,64]
    const float* preds = (const float*)d_in[1];   // [4,4,64,64,64]

    float* num_s    = (float*)d_ws;               // 4 * P3 floats
    float* den_s    = num_s + 4 * P3;             // P3 floats
    float* partials = den_s + P3;                 // 256 * 256 floats

    dim3 g1(PDIM, PDIM / 4, 1);                   // z, y-tiles
    sncut_stencil<<<g1, 256, 0, stream>>>(batch, preds, num_s, den_s);

    sncut_wmma<<<dim3(WAVES2 / 4), 128, 0, stream>>>(preds, num_s, den_s, partials);

    sncut_final<<<dim3(1), 256, 0, stream>>>(partials, (float*)d_out);
}
